// CausalSelfAttention_54047868453458
// MI455X (gfx1250) — compile-verified, matrix operand order checked by reading, not run on hardware
//
#include <hip/hip_runtime.h>

// ---- CDNA5 WMMA types -------------------------------------------------------
typedef _Float16 v16h __attribute__((ext_vector_type(16)));
typedef _Float16 h8   __attribute__((ext_vector_type(8)));
typedef _Float16 h4   __attribute__((ext_vector_type(4)));
typedef float    v8f  __attribute__((ext_vector_type(8)));

#define BATCH   4
#define SEQ     2048
#define DMODEL  1024
#define NHEADS  16
#define HDIM    64
#define MROWS   (BATCH * SEQ)     // 8192
#define QKV_N   (3 * DMODEL)      // 3072
#define SM_SCALE 0.125f           // 1/sqrt(64)

static __device__ __forceinline__ v8f wmma_f16(v16h a, v16h b, v8f c) {
  return __builtin_amdgcn_wmma_f32_16x16x32_f16(false, a, false, b, (short)0, c,
                                                false, false);
}

static __device__ __forceinline__ v8f vzero8() {
  v8f z = {0.f, 0.f, 0.f, 0.f, 0.f, 0.f, 0.f, 0.f};
  return z;
}

// Fragment = two contiguous 16B chunks at p and p+16 halfs (ISA 16-bit A/B
// layout: element e -> K = (e>>3)*16 + hlf*8 + (e&7), lane supplies hlf*8).
static __device__ __forceinline__ v16h load_frag(const _Float16* p) {
  h8 lo = *(const h8*)(p);
  h8 hi = *(const h8*)(p + 16);
  return __builtin_shufflevector(lo, hi, 0, 1, 2, 3, 4, 5, 6, 7,
                                 8, 9, 10, 11, 12, 13, 14, 15);
}

// ---------------------------------------------------------------------------
// Prep 0a: x (f32) -> x16 (f16), same layout. 4 elements/thread.
// ---------------------------------------------------------------------------
__global__ __launch_bounds__(256) void cvt_f16_kernel(
    const float* __restrict__ in, _Float16* __restrict__ out) {
  size_t i = ((size_t)blockIdx.x * blockDim.x + threadIdx.x) * 4;
  float4 v = *(const float4*)(in + i);
  h4 o = {(_Float16)v.x, (_Float16)v.y, (_Float16)v.z, (_Float16)v.w};
  *(h4*)(out + i) = o;
}

// ---------------------------------------------------------------------------
// Prep 0b: W [K][N] f32 -> Wt [N][K] f16 (LDS 32x32 tile transpose).
// grid = (N/32, K/32), block = 256.
// ---------------------------------------------------------------------------
__global__ __launch_bounds__(256) void transpose_cvt_kernel(
    const float* __restrict__ in, _Float16* __restrict__ out, int K, int N) {
  __shared__ _Float16 tile[32][33];
  const int tx = threadIdx.x & 31;
  const int ty = threadIdx.x >> 5;   // 0..7
  const int k0 = blockIdx.y * 32;
  const int n0 = blockIdx.x * 32;
#pragma unroll
  for (int i = 0; i < 4; ++i)
    tile[ty + i * 8][tx] = (_Float16)in[(size_t)(k0 + ty + i * 8) * N + n0 + tx];
  __syncthreads();
#pragma unroll
  for (int i = 0; i < 4; ++i)
    out[(size_t)(n0 + ty + i * 8) * K + k0 + tx] = tile[tx][ty + i * 8];
}

// ---------------------------------------------------------------------------
// Kernel 1: qkv = x16 @ Wt^T + b ; q,k head-major f16; v stored TRANSPOSED
// [bh][d][s]. One wave -> 32x64 strip (2 A-frags x 4 B-frags = 8 WMMA/step,
// 12 b128 loads/step). grid = (M/32, 3072/256), block = 128 (4 waves).
// ---------------------------------------------------------------------------
__global__ __launch_bounds__(128) void qkv_kernel(
    const _Float16* __restrict__ x16, const _Float16* __restrict__ Wt,
    const float* __restrict__ bias, _Float16* __restrict__ qw,
    _Float16* __restrict__ kw, _Float16* __restrict__ vt) {
  const int lane = threadIdx.x & 31;
  const int wave = threadIdx.x >> 5;
  const int hlf  = lane >> 4;
  const int l16  = lane & 15;
  const int row0 = blockIdx.x * 32;
  const int col0 = blockIdx.y * 256 + wave * 64;

  v8f acc[2][4];
#pragma unroll
  for (int rr = 0; rr < 2; ++rr)
#pragma unroll
    for (int t = 0; t < 4; ++t) acc[rr][t] = vzero8();

  for (int k0 = 0; k0 < DMODEL; k0 += 32) {
    v16h a[2];
#pragma unroll
    for (int rr = 0; rr < 2; ++rr)
      a[rr] = load_frag(x16 + (size_t)(row0 + rr * 16 + l16) * DMODEL + k0 + hlf * 8);
#pragma unroll
    for (int t = 0; t < 4; ++t) {
      v16h b = load_frag(Wt + (size_t)(col0 + t * 16 + l16) * DMODEL + k0 + hlf * 8);
      acc[0][t] = wmma_f16(a[0], b, acc[0][t]);
      acc[1][t] = wmma_f16(a[1], b, acc[1][t]);
    }
  }

#pragma unroll
  for (int rr = 0; rr < 2; ++rr) {
#pragma unroll
    for (int t = 0; t < 4; ++t) {
#pragma unroll
      for (int r = 0; r < 8; ++r) {
        int row = row0 + rr * 16 + r + hlf * 8;
        int c   = col0 + t * 16 + l16;
        float val = acc[rr][t][r] + bias[c];
        int which = c >> 10;              // 0=q 1=k 2=v
        int h  = (c & 1023) >> 6;
        int d  = c & 63;
        int bi = row >> 11;
        int s  = row & 2047;
        int bh = bi * NHEADS + h;
        if (which == 2) {
          vt[((size_t)bh * HDIM + d) * SEQ + s] = (_Float16)val;   // transposed
        } else {
          _Float16* dst = (which == 0) ? qw : kw;
          dst[((size_t)bh * SEQ + s) * HDIM + d] = (_Float16)val;
        }
      }
    }
  }
}

// ---------------------------------------------------------------------------
// Kernel 2: causal flash attention. One wave per (b,h, 16-row q tile).
// All fragments (Q, K^T, V^T, P-from-LDS) are pairs of b128 loads.
// grid = 1024 blocks x 256 threads (8 waves) = 8192 tiles.
// ---------------------------------------------------------------------------
__global__ __launch_bounds__(256) void attn_kernel(
    const _Float16* __restrict__ qw, const _Float16* __restrict__ kw,
    const _Float16* __restrict__ vt, _Float16* __restrict__ cw) {
  __shared__ _Float16 plds[8 * 16 * 32];
  const int lane = threadIdx.x & 31;
  const int wave = threadIdx.x >> 5;
  const int hlf  = lane >> 4;
  const int l16  = lane & 15;
  _Float16* pbuf = plds + wave * 512;

  const int tile  = blockIdx.x * 8 + wave;
  const int bh    = tile >> 7;
  const int qt    = tile & 127;
  const int qbase = qt * 16;
  const _Float16* qh  = qw + (size_t)bh * SEQ * HDIM;
  const _Float16* kh  = kw + (size_t)bh * SEQ * HDIM;
  const _Float16* vth = vt + (size_t)bh * HDIM * SEQ;

  v16h qa[2];
#pragma unroll
  for (int j = 0; j < 2; ++j)
    qa[j] = load_frag(qh + (size_t)(qbase + l16) * HDIM + j * 32 + hlf * 8);

  v8f acc[4];
#pragma unroll
  for (int t = 0; t < 4; ++t) acc[t] = vzero8();
  float m_i[8], l_i[8];
#pragma unroll
  for (int r = 0; r < 8; ++r) { m_i[r] = -1e30f; l_i[r] = 0.f; }

  for (int kb = 0; kb < qbase + 16; kb += 32) {
    v8f s0 = vzero8(), s1 = vzero8();
#pragma unroll
    for (int j = 0; j < 2; ++j) {
      v16h b0 = load_frag(kh + (size_t)(kb + l16) * HDIM + j * 32 + hlf * 8);
      v16h b1 = load_frag(kh + (size_t)(kb + 16 + l16) * HDIM + j * 32 + hlf * 8);
      s0 = wmma_f16(qa[j], b0, s0);
      s1 = wmma_f16(qa[j], b1, s1);
    }

#pragma unroll
    for (int r = 0; r < 8; ++r) {
      int qrow = qbase + r + hlf * 8;
      float v0 = (kb + l16 <= qrow)      ? s0[r] * SM_SCALE : -1e30f;
      float v1 = (kb + 16 + l16 <= qrow) ? s1[r] * SM_SCALE : -1e30f;
      float bm = fmaxf(v0, v1);
#pragma unroll
      for (int off = 1; off < 16; off <<= 1) bm = fmaxf(bm, __shfl_xor(bm, off, 32));
      float mnew  = fmaxf(m_i[r], bm);
      float alpha = __expf(m_i[r] - mnew);
      m_i[r] = mnew;
      float p0 = __expf(v0 - mnew);
      float p1 = __expf(v1 - mnew);
      float rs = p0 + p1;
#pragma unroll
      for (int off = 1; off < 16; off <<= 1) rs += __shfl_xor(rs, off, 32);
      l_i[r] = l_i[r] * alpha + rs;
#pragma unroll
      for (int t = 0; t < 4; ++t) acc[t][r] *= alpha;
      int m = r + hlf * 8;
      pbuf[m * 32 + l16]      = (_Float16)p0;
      pbuf[m * 32 + 16 + l16] = (_Float16)p1;
    }
    asm volatile("s_wait_dscnt 0" ::: "memory");   // intra-wave LDS RAW only

    v16h pa = load_frag(pbuf + l16 * 32 + hlf * 8);

#pragma unroll
    for (int t = 0; t < 4; ++t) {
      v16h bv = load_frag(vth + (size_t)(t * 16 + l16) * SEQ + kb + hlf * 8);
      acc[t] = wmma_f16(pa, bv, acc[t]);
    }
  }

  const int bi = bh >> 4, h = bh & 15;
#pragma unroll
  for (int r = 0; r < 8; ++r) {
    float inv = 1.0f / l_i[r];
    int srow = qbase + r + hlf * 8;
#pragma unroll
    for (int t = 0; t < 4; ++t) {
      size_t idx = (size_t)(bi * SEQ + srow) * DMODEL + h * HDIM + t * 16 + l16;
      cw[idx] = (_Float16)(acc[t][r] * inv);
    }
  }
}

// ---------------------------------------------------------------------------
// Kernel 3: out = ctx @ W_proj + b_proj. Same 32x64-per-wave structure.
// grid = (M/32, 1024/256), block = 128.
// ---------------------------------------------------------------------------
__global__ __launch_bounds__(128) void proj_kernel(
    const _Float16* __restrict__ cx, const _Float16* __restrict__ Wt,
    const float* __restrict__ bias, float* __restrict__ out) {
  const int lane = threadIdx.x & 31;
  const int wave = threadIdx.x >> 5;
  const int hlf  = lane >> 4;
  const int l16  = lane & 15;
  const int row0 = blockIdx.x * 32;
  const int col0 = blockIdx.y * 256 + wave * 64;

  v8f acc[2][4];
#pragma unroll
  for (int rr = 0; rr < 2; ++rr)
#pragma unroll
    for (int t = 0; t < 4; ++t) acc[rr][t] = vzero8();

  for (int k0 = 0; k0 < DMODEL; k0 += 32) {
    v16h a[2];
#pragma unroll
    for (int rr = 0; rr < 2; ++rr)
      a[rr] = load_frag(cx + (size_t)(row0 + rr * 16 + l16) * DMODEL + k0 + hlf * 8);
#pragma unroll
    for (int t = 0; t < 4; ++t) {
      v16h b = load_frag(Wt + (size_t)(col0 + t * 16 + l16) * DMODEL + k0 + hlf * 8);
      acc[0][t] = wmma_f16(a[0], b, acc[0][t]);
      acc[1][t] = wmma_f16(a[1], b, acc[1][t]);
    }
  }

#pragma unroll
  for (int rr = 0; rr < 2; ++rr)
#pragma unroll
    for (int t = 0; t < 4; ++t)
#pragma unroll
      for (int r = 0; r < 8; ++r) {
        int row = row0 + rr * 16 + r + hlf * 8;
        int c   = col0 + t * 16 + l16;
        out[(size_t)row * DMODEL + c] = acc[rr][t][r] + bias[c];
      }
}

// ---------------------------------------------------------------------------
extern "C" void kernel_launch(void* const* d_in, const int* in_sizes, int n_in,
                              void* d_out, int out_size, void* d_ws, size_t ws_size,
                              hipStream_t stream) {
  const float* x      = (const float*)d_in[0];
  const float* W_attn = (const float*)d_in[1];
  const float* b_attn = (const float*)d_in[2];
  const float* W_proj = (const float*)d_in[3];
  const float* b_proj = (const float*)d_in[4];
  float* out = (float*)d_out;

  // ws carve (halfs): x16 8M | Wat 3M | Wpt 1M | q 8M | k 8M | v 8M | ctx 8M
  _Float16* x16 = (_Float16*)d_ws;
  _Float16* Wat = x16 + (size_t)MROWS * DMODEL;            // 8388608
  _Float16* Wpt = Wat + (size_t)DMODEL * QKV_N;            // +3145728
  _Float16* qw  = Wpt + (size_t)DMODEL * DMODEL;           // +1048576
  const size_t headElems = (size_t)BATCH * NHEADS * SEQ * HDIM;
  _Float16* kw = qw + headElems;
  _Float16* vt = kw + headElems;
  _Float16* cw = vt + headElems;

  // prep: f16 copies / transposes
  cvt_f16_kernel<<<(MROWS * DMODEL / 4) / 256, 256, 0, stream>>>(x, x16);
  {
    dim3 g(QKV_N / 32, DMODEL / 32);
    transpose_cvt_kernel<<<g, 256, 0, stream>>>(W_attn, Wat, DMODEL, QKV_N);
  }
  {
    dim3 g(DMODEL / 32, DMODEL / 32);
    transpose_cvt_kernel<<<g, 256, 0, stream>>>(W_proj, Wpt, DMODEL, DMODEL);
  }

  dim3 g1(MROWS / 32, QKV_N / 256);
  qkv_kernel<<<g1, 128, 0, stream>>>(x16, Wat, b_attn, qw, kw, vt);

  attn_kernel<<<(BATCH * NHEADS * (SEQ / 16)) / 8, 256, 0, stream>>>(qw, kw, vt, cw);

  dim3 g3(MROWS / 32, DMODEL / 256);
  proj_kernel<<<g3, 128, 0, stream>>>(cw, Wpt, b_proj, out);
}
